// Flow_SNet_46797963657899
// MI455X (gfx1250) — matrix operand order repaired and checked by reading, not compiled, not run on hardware
//
#include <hip/hip_runtime.h>

typedef _Float16 half_t;
typedef __attribute__((ext_vector_type(16))) _Float16     v16h;
typedef __attribute__((ext_vector_type(8)))  float        v8f;
typedef __attribute__((ext_vector_type(4)))  unsigned int u32x4;

union AFrag { v16h v; u32x4 q[2]; half_t h[16]; };
union CFrag { v8f v; float f[8]; };
union Stg8  { half_t h[8]; u32x4 v; };

// ---------------------------------------------------------------------------
// Weight pre-pack: f32 [Cout][Ktot] -> f16 [Cp128][Kp], zero padded so the
// conv loop needs no guards/converts on the B side.
// ---------------------------------------------------------------------------
__global__ void pack_weights_kernel(const float* __restrict__ w, half_t* __restrict__ wp,
                                    int Cout, int Ktot, int Kp, long total)
{
    long i = (long)blockIdx.x * blockDim.x + threadIdx.x;
    if (i >= total) return;
    int k  = (int)(i % Kp);
    int co = (int)(i / Kp);
    float v = (co < Cout && k < Ktot) ? w[(long)co * Ktot + k] : 0.0f;
    wp[i] = (half_t)v;
}

// ---------------------------------------------------------------------------
// Implicit-GEMM 3D conv. Block = 256 thr = 8 waves; tile = 64 voxels x 128
// output channels (4 M-subtiles per wave). Per K-step of 32: cooperative
// im2col stage (f32->f16, hoisted decode + magic-div + LDS tap table) into
// LDS, then each wave issues four v_wmma_f32_16x16x32_f16 sharing one
// packed-f16 B fragment (4x B reuse, barriers amortized over 4 MMAs).
// ---------------------------------------------------------------------------
__global__ __launch_bounds__(256) void conv3d_wmma_kernel(
    const float* __restrict__ in1, int Cin1,
    const float* __restrict__ in2, int Cin2,
    const half_t* __restrict__ wp, const float* __restrict__ bias,
    float* __restrict__ out,
    int ID, int IH, int IW,
    int OD, int OH, int OW, int Cout,
    int KH, int KW,
    int sz, int sy, int sx,
    int pz, int py, int px,
    int taps, unsigned tmagic, int Ktot, int Kp,
    int leaky)
{
    const int b    = blockIdx.x;
    const int mtb  = blockIdx.y;
    const int ct   = blockIdx.z;
    const int tid  = threadIdx.x;
    const int lane = tid & 31;
    const int wave = tid >> 5;

    const int S      = OD * OH * OW;
    const int ISize  = ID * IH * IW;
    const int s_base = mtb * 64;
    const int hi     = lane >> 4;
    const int mrow   = lane & 15;
    const int co     = ct * 128 + wave * 16 + (lane & 15);

    __shared__ __align__(16) half_t aT[64 * 32];   // [m 0..63][k 0..31], 4KB
    __shared__ int tapOff[32];

    if (tid < taps) {
        int kz = tid / (KH * KW); int r = tid - kz * KH * KW;
        int ky = r / KW;          int kx = r - ky * KW;
        tapOff[tid] = (kz << 16) | (ky << 8) | kx;
    }

    // ---- hoisted staging decode: this thread always stages aT[m][k0..k0+7]
    const int m  = tid >> 2;          // 0..63
    const int k0 = (tid & 3) * 8;     // 0,8,16,24
    const int s  = s_base + m;
    const bool sOK = s < S;
    const int ss = sOK ? s : 0;
    int oz = ss / (OH * OW); int r0 = ss - oz * (OH * OW);
    int oy = r0 / OW;        int ox = r0 - oy * OW;
    const int izb = oz * sz - pz;
    const int iyb = oy * sy - py;
    const int ixb = ox * sx - px;
    const long base1 = (long)b * Cin1 * ISize;
    const long base2 = (long)b * Cin2 * ISize;

    v8f acc[4];
    #pragma unroll
    for (int mi = 0; mi < 4; ++mi)
        acc[mi] = (v8f){0.f, 0.f, 0.f, 0.f, 0.f, 0.f, 0.f, 0.f};

    const half_t* wrow = wp + (long)co * Kp + hi * 16;  // padded: never OOB

    for (int kb = 0; kb < Ktot; kb += 32) {
        __syncthreads();
        // ---- stage 8 contiguous halves of the im2col tile ----
        Stg8 stg;
        #pragma unroll
        for (int j = 0; j < 8; ++j) {
            int kk = kb + k0 + j;
            unsigned ci = (unsigned)(((unsigned long long)(unsigned)kk * tmagic) >> 32);
            int t = kk - (int)ci * taps;            // exact: magic div is exact here
            int off = tapOff[t];
            int iz = izb + (off >> 16);
            int iy = iyb + ((off >> 8) & 255);
            int ix = ixb + (off & 255);
            float v = 0.0f;
            if (sOK && kk < Ktot &&
                (unsigned)iz < (unsigned)ID &&
                (unsigned)iy < (unsigned)IH &&
                (unsigned)ix < (unsigned)IW) {
                long sp = ((long)iz * IH + iy) * IW + ix;
                v = (ci < (unsigned)Cin1)
                        ? in1[base1 + (long)ci * ISize + sp]
                        : in2[base2 + (long)(ci - Cin1) * ISize + sp];
            }
            stg.h[j] = (half_t)v;
        }
        *(u32x4*)(aT + m * 32 + k0) = stg.v;        // one ds_store_b128
        __syncthreads();

        // ---- B fragment: two contiguous 16B loads of packed f16 weights ----
        __builtin_prefetch(wrow + kb + 64, 0, 1);
        AFrag Bf;
        const u32x4* bp = (const u32x4*)(wrow + kb);
        Bf.q[0] = bp[0];
        Bf.q[1] = bp[1];

        // ---- 4 A fragments (contiguous k = 16*hi + s, same as B), shared B ----
        #pragma unroll
        for (int mi = 0; mi < 4; ++mi) {
            AFrag A;
            const u32x4* ap = (const u32x4*)(aT + (mi * 16 + mrow) * 32 + hi * 16);
            A.q[0] = ap[0];
            A.q[1] = ap[1];
            acc[mi] = __builtin_amdgcn_wmma_f32_16x16x32_f16(
                false, A.v, false, Bf.v, (short)0, acc[mi], false, false);
        }
    }

    if (co < Cout) {
        float bv = bias[co];
        const long obase = ((long)b * Cout + co) * S;
        #pragma unroll
        for (int mi = 0; mi < 4; ++mi) {
            CFrag R; R.v = acc[mi];
            #pragma unroll
            for (int i = 0; i < 8; ++i) {
                int s0 = s_base + mi * 16 + i + hi * 8;   // ISA C/D layout
                if (s0 < S) {
                    float v = R.f[i] + bv;
                    if (leaky) v = v > 0.0f ? v : 0.01f * v;
                    out[obase + s0] = v;
                }
            }
        }
    }
}

// ---------------------------------------------------------------------------
// Aux elementwise kernels (resize / warp / splat / normalize / flow / output)
// ---------------------------------------------------------------------------
__device__ inline float tri_sample(const float* __restrict__ f, int D, int H, int W,
                                   float pz, float py, float px)
{
    int z0 = (int)floorf(pz), y0 = (int)floorf(py), x0 = (int)floorf(px);
    float fz = pz - z0, fy = py - y0, fx = px - x0;
    int z1 = z0 + 1, y1 = y0 + 1, x1 = x0 + 1;
    z0 = min(max(z0, 0), D - 1); z1 = min(max(z1, 0), D - 1);
    y0 = min(max(y0, 0), H - 1); y1 = min(max(y1, 0), H - 1);
    x0 = min(max(x0, 0), W - 1); x1 = min(max(x1, 0), W - 1);
    float c000 = f[((long)z0 * H + y0) * W + x0], c001 = f[((long)z0 * H + y0) * W + x1];
    float c010 = f[((long)z0 * H + y1) * W + x0], c011 = f[((long)z0 * H + y1) * W + x1];
    float c100 = f[((long)z1 * H + y0) * W + x0], c101 = f[((long)z1 * H + y0) * W + x1];
    float c110 = f[((long)z1 * H + y1) * W + x0], c111 = f[((long)z1 * H + y1) * W + x1];
    float c00 = c000 + (c001 - c000) * fx, c01 = c010 + (c011 - c010) * fx;
    float c10 = c100 + (c101 - c100) * fx, c11 = c110 + (c111 - c110) * fx;
    float c0 = c00 + (c01 - c00) * fy, c1 = c10 + (c11 - c10) * fy;
    return c0 + (c1 - c0) * fz;
}

__global__ void resize_tri_kernel(const float* __restrict__ src, float* __restrict__ dst,
                                  int B, int C, int sd, int sh, int sw,
                                  int dd, int dh, int dw,
                                  float vz, float vy, float vx, int isflow)
{
    long total = (long)B * C * dd * dh * dw;
    long i = (long)blockIdx.x * blockDim.x + threadIdx.x;
    if (i >= total) return;
    int x = (int)(i % dw); long t = i / dw;
    int y = (int)(t % dh); t /= dh;
    int z = (int)(t % dd); t /= dd;
    int c = (int)(t % C);  int b = (int)(t / C);
    float pz = (z + 0.5f) * ((float)sd / dd) - 0.5f;
    float py = (y + 0.5f) * ((float)sh / dh) - 0.5f;
    float px = (x + 0.5f) * ((float)sw / dw) - 0.5f;
    const float* s = src + ((long)b * C + c) * ((long)sd * sh * sw);
    float v = tri_sample(s, sd, sh, sw, pz, py, px);
    if (isflow) v *= (c == 0 ? vz : (c == 1 ? vy : vx));
    dst[i] = v;
}

__global__ void warp_kernel(const float* __restrict__ feat, long featBstride,
                            const float* __restrict__ flow, float* __restrict__ out,
                            int B, int C, int D, int H, int W)
{
    long DHW = (long)D * H * W;
    long total = (long)B * C * DHW;
    long i = (long)blockIdx.x * blockDim.x + threadIdx.x;
    if (i >= total) return;
    long sp = i % DHW; long t = i / DHW;
    int c = (int)(t % C); int b = (int)(t / C);
    int x = (int)(sp % W); long t2 = sp / W;
    int y = (int)(t2 % H); int z = (int)(t2 / H);
    const float* fl = flow + (long)b * 3 * DHW;
    float pz = z + fl[0 * DHW + sp];
    float py = y + fl[1 * DHW + sp];
    float px = x + fl[2 * DHW + sp];
    const float* fb = feat + featBstride * b + (long)c * DHW;
    out[i] = tri_sample(fb, D, H, W, pz, py, px);
}

__global__ void splat_kernel(const float* __restrict__ feat, const float* __restrict__ flow,
                             float* __restrict__ out, int B, int C,
                             int sd, int sh, int sw, int dd, int dh, int dw)
{
    long S = (long)sd * sh * sw;
    long total = (long)B * S;
    long i = (long)blockIdx.x * blockDim.x + threadIdx.x;
    if (i >= total) return;
    long sp = i % S; int b = (int)(i / S);
    int x = (int)(sp % sw); long t = sp / sw;
    int y = (int)(t % sh);  int z = (int)(t / sh);
    long P = (long)dd * dh * dw;
    const float* fl = flow + (long)b * 3 * S;
    float pz = (z + fl[0 * S + sp]) * ((float)dd / sd);
    float py = (y + fl[1 * S + sp]) * ((float)dh / sh);
    float px = (x + fl[2 * S + sp]) * ((float)dw / sw);
    int z0 = (int)floorf(pz), y0 = (int)floorf(py), x0 = (int)floorf(px);
    float fz = pz - z0, fy = py - y0, fx = px - x0;
    #pragma unroll
    for (int dz = 0; dz < 2; ++dz)
    #pragma unroll
    for (int dy = 0; dy < 2; ++dy)
    #pragma unroll
    for (int dx = 0; dx < 2; ++dx) {
        int iz = z0 + dz, iy = y0 + dy, ix = x0 + dx;
        if (iz < 0 || iz >= dd || iy < 0 || iy >= dh || ix < 0 || ix >= dw) continue;
        float w = (dz ? fz : 1.f - fz) * (dy ? fy : 1.f - fy) * (dx ? fx : 1.f - fx);
        long lin = ((long)iz * dh + iy) * dw + ix;
        for (int c = 0; c < C; ++c)
            atomicAdd(&out[(long)c * P + lin], w * feat[((long)b * C + c) * S + sp]);
        atomicAdd(&out[(long)C * P + lin], w);   // homogeneous mask channel
    }
}

__global__ void maxmask_kernel(const float* __restrict__ mask, long P,
                               unsigned int* __restrict__ dmax)
{
    long i = (long)blockIdx.x * blockDim.x + threadIdx.x;
    if (i >= P) return;
    float v = mask[i]; if (v < 0.f) v = 0.f;   // non-negative: uint order == float order
    atomicMax(dmax, __float_as_uint(v));
}

__global__ void normsplat_kernel(float* __restrict__ sp, int C, long P,
                                 const unsigned int* __restrict__ dmax)
{
    long total = (long)C * P;
    long i = (long)blockIdx.x * blockDim.x + threadIdx.x;
    if (i >= total) return;
    float m = __uint_as_float(*dmax);
    m = fmaxf(m, 1e-6f);
    sp[i] *= (1.0f / m);
}

__global__ void flowadd_kernel(float* __restrict__ flow, const float* __restrict__ delta,
                               int B, long S)
{
    long total = (long)B * 3 * S;
    long i = (long)blockIdx.x * blockDim.x + threadIdx.x;
    if (i >= total) return;
    long sp = i % S; long t = i / S;
    int c = (int)(t % 3); int b = (int)(t / 3);
    flow[((long)b * 3 + c) * S + sp] += delta[((long)b * 4 + c) * S + sp];
}

__global__ void finalize_kernel(const float* __restrict__ flow, float* __restrict__ out, long P)
{
    long total = 9 * P;
    long i = (long)blockIdx.x * blockDim.x + threadIdx.x;
    if (i >= total) return;
    long sp = i % P; long t = i / P;
    int n = (int)(t % 3); int c = (int)(t / 3);
    out[((long)c * 3 + n) * P + sp] = flow[((long)n * 3 + (2 - c)) * P + sp];
}

// ---------------------------------------------------------------------------
// Host orchestration
// ---------------------------------------------------------------------------
extern "C" void kernel_launch(void* const* d_in, const int* in_sizes, int n_in,
                              void* d_out, int out_size, void* d_ws, size_t ws_size,
                              hipStream_t stream)
{
    (void)in_sizes; (void)n_in; (void)out_size; (void)ws_size;

    const float* xs = (const float*)d_in[0];   // (3,1,64,64,64) after moveaxis fold
    const float* w_enc[4], *b_enc[4];
    for (int d = 0; d < 4; ++d) {
        w_enc[d] = (const float*)d_in[1 + 2 * d];
        b_enc[d] = (const float*)d_in[2 + 2 * d];
    }
    const float* w_dec[3], *b_dec[3], *w3[3], *b3[3], *w_flo[3], *b_flo[3];
    for (int u = 0; u < 3; ++u) {
        w_dec[u] = (const float*)d_in[9  + 6 * u];
        b_dec[u] = (const float*)d_in[10 + 6 * u];
        w3[u]    = (const float*)d_in[11 + 6 * u];
        b3[u]    = (const float*)d_in[12 + 6 * u];
        w_flo[u] = (const float*)d_in[13 + 6 * u];
        b_flo[u] = (const float*)d_in[14 + 6 * u];
    }

    const int CH[4] = {16, 32, 64, 128};
    const int UD[4] = {64, 32, 16, 8};     // slab grid (H never pooled)
    const int UH[4] = {64, 64, 64, 64};
    const int UW[4] = {64, 32, 16, 8};
    const int I3[4] = {64, 32, 16, 8};     // isotropic grid
    long SU[4], P3[4];
    for (int d = 0; d < 4; ++d) {
        SU[d] = (long)UD[d] * UH[d] * UW[d];
        P3[d] = (long)I3[d] * I3[d] * I3[d];
    }

    float* Wb = (float*)d_ws;
    size_t off = 0;
    auto alloc = [&](size_t nfloats) -> float* {
        float* p = Wb + off; off += (nfloats + 63) & ~(size_t)63; return p;
    };
    auto allocH = [&](size_t nhalf) -> half_t* {
        return (half_t*)alloc((nhalf + 1) / 2);
    };

    const int BLK = 256;
    auto ew = [&](long total) { return dim3((unsigned)((total + BLK - 1) / BLK)); };

    // ---- pre-pack all 13 conv weight tensors to padded f16 ----
    struct Packed { half_t* wp; int Kp; };
    auto prep = [&](const float* w, int Cout, int Ktot) -> Packed {
        int Kp = (Ktot + 31) & ~31;
        int Cp = ((Cout + 127) / 128) * 128;
        long total = (long)Cp * Kp;
        half_t* wp = allocH((size_t)total);
        pack_weights_kernel<<<ew(total), BLK, 0, stream>>>(w, wp, Cout, Ktot, Kp, total);
        return {wp, Kp};
    };
    Packed p_enc[4], p_dec[3], p_w3[3], p_flo[3];
    {
        int inc = 1;
        for (int d = 0; d < 4; ++d) {
            p_enc[d] = prep(w_enc[d], CH[d], inc * 9);
            inc = CH[d];
        }
        for (int u = 0; u < 3; ++u) {
            p_dec[u] = prep(w_dec[u], CH[u], (CH[u + 1] + CH[u]) * 9);
            p_w3[u]  = prep(w3[u],    CH[u], (CH[u + 1] + CH[u]) * 27);
            p_flo[u] = prep(w_flo[u], 4,     (2 * CH[u]) * 27);
        }
    }

    float* skip[4];
    skip[0] = alloc(3L * 16  * SU[0]);
    skip[1] = alloc(3L * 32  * SU[1]);
    skip[2] = alloc(3L * 64  * SU[2]);
    skip[3] = alloc(3L * 128 * SU[3]);
    float* curA  = alloc(3L * 16 * SU[0]);
    float* curB  = alloc(3L * 16 * SU[0]);
    float* x3A   = alloc(16L * P3[0]);
    float* x3B   = alloc(16L * P3[0]);
    float* flowA = alloc(9L * SU[0]);
    float* flowB = alloc(9L * SU[0]);
    unsigned int* dmax = (unsigned int*)alloc(16);
    const size_t arena_base = off;

    auto conv = [&](const float* in1, int Cin1, const float* in2, int Cin2,
                    Packed pk, const float* bs, float* outp, int B,
                    int ID, int IH, int IW, int OD, int OH, int OW, int Cout,
                    int kd, int kh, int kw, int s0, int s1, int s2,
                    int p0, int p1, int p2, int leaky) {
        int taps = kd * kh * kw;
        int Ktot = (Cin1 + Cin2) * taps;
        unsigned magic = (unsigned)((0x100000000ULL + (unsigned)taps - 1) / (unsigned)taps);
        long S = (long)OD * OH * OW;
        dim3 g((unsigned)B, (unsigned)((S + 63) / 64), (unsigned)((Cout + 127) / 128));
        conv3d_wmma_kernel<<<g, BLK, 0, stream>>>(in1, Cin1, in2, Cin2, pk.wp, bs, outp,
            ID, IH, IW, OD, OH, OW, Cout, kh, kw, s0, s1, s2, p0, p1, p2,
            taps, magic, Ktot, pk.Kp, leaky);
    };

    // ---------------- encoder (slab net, kernel (3,1,3)) ----------------
    conv(xs,      1,  nullptr, 0, p_enc[0], b_enc[0], skip[0], 3, 64,64,64, 64,64,64, 16,  3,1,3, 1,1,1, 1,0,1, 1);
    conv(skip[0], 16, nullptr, 0, p_enc[1], b_enc[1], skip[1], 3, 64,64,64, 32,64,32, 32,  3,1,3, 2,1,2, 0,0,0, 1);
    conv(skip[1], 32, nullptr, 0, p_enc[2], b_enc[2], skip[2], 3, 32,64,32, 16,64,16, 64,  3,1,3, 2,1,2, 0,0,0, 1);
    conv(skip[2], 64, nullptr, 0, p_enc[3], b_enc[3], skip[3], 3, 16,64,16,  8,64, 8, 128, 3,1,3, 2,1,2, 0,0,0, 1);

    hipMemsetAsync(flowA, 0, (size_t)(9L * SU[3]) * sizeof(float), stream);
    hipMemsetAsync(x3A,   0, (size_t)(128L * P3[3]) * sizeof(float), stream);

    float* cur = skip[3];
    float* x3  = x3A;
    float* flow = flowA;
    float* cur_next  = curA;
    float* x3_next   = x3B;
    float* flow_next = flowB;

    // ---------------- decoder u = 2,1,0 ----------------
    for (int u = 2; u >= 0; --u) {
        const int C = CH[u], C1 = CH[u + 1];
        const int Dp = UD[u + 1], Hp = UH[u + 1], Wp = UW[u + 1];
        const int D = UD[u], H = UH[u], Wd = UW[u];
        const long S = SU[u];
        const int I3u = I3[u], I3p = I3[u + 1];
        const long P = P3[u];

        size_t aoff = arena_base;
        auto aalloc = [&](size_t n) -> float* {
            float* p = Wb + aoff; aoff += (n + 63) & ~(size_t)63; return p;
        };

        // 1. up = resize(cur -> slab level u)
        float* up = aalloc(3L * C1 * S);
        resize_tri_kernel<<<ew(3L * C1 * S), BLK, 0, stream>>>(
            cur, up, 3, C1, Dp, Hp, Wp, D, H, Wd, 0.f, 0.f, 0.f, 0);
        // 2. cur = leaky(conv(concat(up, skip), w_dec))
        float* newcur = cur_next;
        conv(up, C1, skip[u], C, p_dec[u], b_dec[u], newcur, 3,
             D, H, Wd, D, H, Wd, C, 3,1,3, 1,1,1, 1,0,1, 1);
        aoff = arena_base;                         // 'up' dead

        // 3. flow_r = resize_flow(flow -> slab level u) (used by splat, warp, add)
        float* flow_r = flow_next;
        resize_tri_kernel<<<ew(3L * 3 * S), BLK, 0, stream>>>(
            flow, flow_r, 3, 3, Dp, Hp, Wp, D, H, Wd,
            (float)D / Dp, (float)H / Hp, (float)Wd / Wp, 1);

        // 4. sp = splat(skip, flow_r -> isotropic grid), summed over batch via atomics
        float* spbuf = aalloc((long)(C + 1) * P);
        hipMemsetAsync(spbuf, 0, (size_t)((long)(C + 1) * P) * sizeof(float), stream);
        splat_kernel<<<ew(3L * S), BLK, 0, stream>>>(
            skip[u], flow_r, spbuf, 3, C, D, H, Wd, I3u, I3u, I3u);
        // 5. normalize by max of mask channel
        hipMemsetAsync(dmax, 0, sizeof(unsigned int), stream);
        maxmask_kernel<<<ew(P), BLK, 0, stream>>>(spbuf + (long)C * P, P, dmax);
        normsplat_kernel<<<ew((long)C * P), BLK, 0, stream>>>(spbuf, C, P, dmax);

        // 6. up3 = resize(x3 -> isotropic level u)
        float* up3 = aalloc((long)C1 * P);
        resize_tri_kernel<<<ew((long)C1 * P), BLK, 0, stream>>>(
            x3, up3, 1, C1, I3p, I3p, I3p, I3u, I3u, I3u, 0.f, 0.f, 0.f, 0);
        // 7. x3 = leaky(conv3(concat(up3, sp), w3))
        float* newx3 = x3_next;
        conv(up3, C1, spbuf, C, p_w3[u], b3[u], newx3, 1,
             I3u, I3u, I3u, I3u, I3u, I3u, C, 3,3,3, 1,1,1, 1,1,1, 1);
        aoff = arena_base;                         // spbuf, up3 dead

        // 8. x3r = resize(x3 -> slab grid); 9. xw = warp(broadcast x3r, flow_r)
        float* x3r = aalloc((long)C * S);
        resize_tri_kernel<<<ew((long)C * S), BLK, 0, stream>>>(
            newx3, x3r, 1, C, I3u, I3u, I3u, D, H, Wd, 0.f, 0.f, 0.f, 0);
        float* xw = aalloc(3L * C * S);
        warp_kernel<<<ew(3L * C * S), BLK, 0, stream>>>(
            x3r, (long)0, flow_r, xw, 3, C, D, H, Wd);

        // 10. delta = conv(concat(cur, xw), w_flo)  (4 ch, no activation)
        float* delta = aalloc(3L * 4 * S);
        conv(newcur, C, xw, C, p_flo[u], b_flo[u], delta, 3,
             D, H, Wd, D, H, Wd, 4, 3,3,3, 1,1,1, 1,1,1, 0);
        // 11. flow = flow_r + delta[:, :3]  (in place on flow_r)
        flowadd_kernel<<<ew(3L * 3 * S), BLK, 0, stream>>>(flow_r, delta, 3, S);

        cur = newcur;  cur_next  = (newcur == curA) ? curB : curA;
        x3  = newx3;   x3_next   = (newx3 == x3A)  ? x3B  : x3A;
        flow = flow_r; flow_next = (flow_r == flowA) ? flowB : flowA;
    }

    // out[b=0][c][n][...] = flow[n][2-c][...]
    finalize_kernel<<<ew(9L * SU[0]), BLK, 0, stream>>>(flow, (float*)d_out, SU[0]);
}
